// BiMamba2Block_37469294690847
// MI455X (gfx1250) — matrix operand order repaired and checked by reading, not compiled, not run on hardware
//
#include <hip/hip_runtime.h>
#include <math.h>

// ---------------- problem constants ----------------
#define BATCH   2
#define SEQL    2048
#define DMODEL  1024
#define DINNER  2048
#define DSTATE  128
#define DCONV   4
#define HEADDIM 64
#define NHEADS  32
#define CONVDIM 2304          // DINNER + 2*DSTATE
#define DINPROJ 4384          // 2*DINNER + 2*DSTATE + NHEADS
#define QCHUNK  128
#define NCHUNK  (SEQL / QCHUNK)   // 16
#define BL      (BATCH * SEQL)    // 4096

typedef __attribute__((ext_vector_type(16))) __bf16 bf16x16;
typedef __attribute__((ext_vector_type(8)))  float  floatx8;
typedef __attribute__((ext_vector_type(4)))  unsigned int u32x4;
typedef __attribute__((ext_vector_type(8)))  int i32x8;
typedef __attribute__((ext_vector_type(4)))  int i32x4;

// vector-of-4-int type matching the async-to-LDS builtin's pointee type
typedef int v4i __attribute__((vector_size(4 * sizeof(int))));
typedef __attribute__((address_space(1))) v4i* gv4i_ptr;
typedef __attribute__((address_space(3))) v4i* lv4i_ptr;

// ---------------- CDNA5 feature probes ----------------
#if __has_builtin(__builtin_amdgcn_global_load_async_to_lds_b128)
#define HAVE_ASYNC_LDS 1
#else
#define HAVE_ASYNC_LDS 0
#endif
#if __has_builtin(__builtin_amdgcn_tensor_load_to_lds)
#define HAVE_TDM 1
#else
#define HAVE_TDM 0
#endif

__device__ __forceinline__ void async_b128(const void* g, void* l) {
#if HAVE_ASYNC_LDS
  __builtin_amdgcn_global_load_async_to_lds_b128((gv4i_ptr)g, (lv4i_ptr)l, 0, 0);
#else
  ((unsigned long long*)l)[0] = ((const unsigned long long*)g)[0];
  ((unsigned long long*)l)[1] = ((const unsigned long long*)g)[1];
#endif
}

__device__ __forceinline__ void wait_async0() {
#if __has_builtin(__builtin_amdgcn_s_wait_asynccnt)
  __builtin_amdgcn_s_wait_asynccnt(0);
#elif HAVE_ASYNC_LDS
  asm volatile("s_wait_asynccnt 0x0" ::: "memory");
#endif
}

__device__ __forceinline__ void wait_tensor0() {
#if __has_builtin(__builtin_amdgcn_s_wait_tensorcnt)
  __builtin_amdgcn_s_wait_tensorcnt(0);
#elif HAVE_TDM
  asm volatile("s_wait_tensorcnt 0x0" ::: "memory");
#endif
}

// ---------------- WMMA helpers (gfx1250, wave32) ----------------
__device__ __forceinline__ floatx8 wmma_bf16(bf16x16 a, bf16x16 b, floatx8 c) {
  return __builtin_amdgcn_wmma_f32_16x16x32_bf16(false, a, false, b, (short)0, c, false, false);
}

// A matrix 16x32 bf16, source stored row-major [row][k].
__device__ __forceinline__ bf16x16 frag_a_rowmajor(const __bf16* p, int stride, int row,
                                                   int khalf, int k0) {
  bf16x16 f;
  const __bf16* r = p + (size_t)row * stride + k0 + 16 * khalf;
#pragma unroll
  for (int i = 0; i < 4; ++i) { f[2*i]   = r[2*i];      f[2*i+1] = r[2*i+1]; }
#pragma unroll
  for (int i = 0; i < 4; ++i) { f[8+2*i] = r[8+2*i];   f[9+2*i] = r[8+2*i+1]; }
  return f;
}

// A matrix, source stored K-major [k][row].
__device__ __forceinline__ bf16x16 frag_a_kmajor(const __bf16* p, int stride, int row,
                                                 int khalf, int k0) {
  bf16x16 f;
  int base = k0 + 16 * khalf;
#pragma unroll
  for (int i = 0; i < 4; ++i) {
    f[2*i]   = p[(size_t)(base + 2*i)     * stride + row];
    f[2*i+1] = p[(size_t)(base + 2*i + 1) * stride + row];
  }
#pragma unroll
  for (int i = 0; i < 4; ++i) {
    f[8+2*i] = p[(size_t)(base + 8 + 2*i)     * stride + row];
    f[9+2*i] = p[(size_t)(base + 8 + 2*i + 1) * stride + row];
  }
  return f;
}

// B matrix, value B[k][col], source stored [col][k] (contiguous in k).
__device__ __forceinline__ bf16x16 frag_b_nmajor(const __bf16* p, int stride, int col,
                                                 int khalf, int k0) {
  bf16x16 f;
  const __bf16* r = p + (size_t)col * stride + k0 + 16 * khalf;
#pragma unroll
  for (int e = 0; e < 16; ++e) f[e] = r[e];
  return f;
}

// B matrix, source stored [k][col] (K-major).
__device__ __forceinline__ bf16x16 frag_b_kmajor(const __bf16* p, int stride, int col,
                                                 int khalf, int k0) {
  bf16x16 f;
  int base = k0 + 16 * khalf;
#pragma unroll
  for (int e = 0; e < 16; ++e) f[e] = p[(size_t)(base + e) * stride + col];
  return f;
}

__device__ __forceinline__ float siluf(float v) { return v / (1.0f + expf(-v)); }
__device__ __forceinline__ float softplusf(float v) {
  return (v > 20.0f) ? v : log1pf(expf(v));
}
__device__ __forceinline__ int memidx(int t, int dir) { return dir ? (SEQL - 1 - t) : t; }

// ---------------- workspace layout (bytes) ----------------
static constexpr size_t OFF_WIN_BF  = 0;
static constexpr size_t OFF_WOUT_BF = OFF_WIN_BF  + (size_t)DINPROJ * DMODEL * 2;
static constexpr size_t OFF_ZX      = OFF_WOUT_BF + (size_t)DMODEL * DINNER * 2;
static constexpr size_t OFF_XS      = OFF_ZX      + (size_t)BL * DINPROJ * 4;
static constexpr size_t OFF_XD      = OFF_XS      + (size_t)BL * DINNER * 4;
static constexpr size_t OFF_BB      = OFF_XD      + (size_t)BL * DINNER * 2;
static constexpr size_t OFF_CB      = OFF_BB      + (size_t)BL * DSTATE * 2;
static constexpr size_t OFF_AB      = OFF_CB      + (size_t)BL * DSTATE * 2;
static constexpr size_t OFF_Y       = OFF_AB      + (size_t)BL * NHEADS * 4;
static constexpr size_t OFF_ST      = OFF_Y       + (size_t)BL * DINNER * 4;
static constexpr size_t OFF_SIN     = OFF_ST      + (size_t)BATCH*NCHUNK*NHEADS*HEADDIM*DSTATE*4;
static constexpr size_t OFF_AL      = OFF_SIN     + (size_t)BATCH*NCHUNK*NHEADS*HEADDIM*DSTATE*2;
static constexpr size_t OFF_YG      = OFF_AL      + (size_t)BATCH*NHEADS*NCHUNK*4 + 256;
static constexpr size_t OFF_YC      = OFF_YG      + (size_t)BL * DINNER * 4;
static constexpr size_t OFF_XBF     = OFF_YC      + (size_t)BL * DINNER * 4;   // BL*1024 bf16
static constexpr size_t OFF_YCBF    = OFF_XBF     + (size_t)BL * DMODEL * 2;   // BL*2048 bf16

// ---------------- kernels ----------------

__global__ __launch_bounds__(256) void k_cvt_bf16(const float* __restrict__ src,
                                                  __bf16* __restrict__ dst, int n) {
  int i = blockIdx.x * 256 + threadIdx.x;
  if (i < n) dst[i] = (__bf16)src[i];
}

// C[M,N] = A[M,K](bf16) x W[N,K](bf16)^T. 128x128 block, 8 waves,
// double-buffered async global->LDS tile pipeline.
__global__ __launch_bounds__(256) void k_gemm_bf16(const __bf16* __restrict__ A,
                                                   const __bf16* __restrict__ W,
                                                   float* __restrict__ C,
                                                   int M, int N, int K) {
  __shared__ __bf16 As[2][128 * 32];
  __shared__ __bf16 Ws[2][128 * 32];
  int tid = threadIdx.x, lane = tid & 31, wave = tid >> 5;
  int lr = lane & 15, khalf = lane >> 4;
  int wm = wave & 3, wn = wave >> 2;             // 4 x 2 wave grid
  int bm = blockIdx.y * 128, bn = blockIdx.x * 128;
  floatx8 acc[2][4] = {};

  auto load_tile = [&](int buf, int k0) {
#pragma unroll
    for (int u = 0; u < 2; ++u) {
      int chunk = tid * 2 + u;                   // 512 b128 chunks per tile
      int row = chunk >> 2;
      int seg = (chunk & 3) * 8;                 // 8 bf16 per b128
      async_b128(A + (size_t)(bm + row) * K + k0 + seg, &As[buf][row * 32 + seg]);
      int wrow = bn + row;
      if (wrow < N) {
        async_b128(W + (size_t)wrow * K + k0 + seg, &Ws[buf][row * 32 + seg]);
      } else {
#pragma unroll
        for (int e = 0; e < 8; ++e) Ws[buf][row * 32 + seg + e] = (__bf16)0.0f;
      }
    }
  };

  load_tile(0, 0);
  int buf = 0;
  for (int k0 = 0; k0 < K; k0 += 32) {
    wait_async0();
    __syncthreads();
    if (k0 + 32 < K) load_tile(buf ^ 1, k0 + 32);

    const __bf16* Ab = &As[buf][0];
    const __bf16* Wb = &Ws[buf][0];
    bf16x16 af[2];
#pragma unroll
    for (int mi = 0; mi < 2; ++mi)
      af[mi] = frag_a_rowmajor(Ab, 32, wm * 32 + mi * 16 + lr, khalf, 0);
#pragma unroll
    for (int ni = 0; ni < 4; ++ni) {
      bf16x16 bf = frag_b_nmajor(Wb, 32, wn * 64 + ni * 16 + lr, khalf, 0);
#pragma unroll
      for (int mi = 0; mi < 2; ++mi) acc[mi][ni] = wmma_bf16(af[mi], bf, acc[mi][ni]);
    }
    buf ^= 1;
  }
#pragma unroll
  for (int mi = 0; mi < 2; ++mi)
#pragma unroll
    for (int ni = 0; ni < 4; ++ni)
#pragma unroll
      for (int r = 0; r < 8; ++r) {
        int rrow = bm + wm * 32 + mi * 16 + r + 8 * khalf;
        int ccol = bn + wn * 64 + ni * 16 + lr;
        if (ccol < N) C[(size_t)rrow * N + ccol] = acc[mi][ni][r];
      }
}

// Per (b,t): depthwise causal conv (direction order) + SiLU, dt softplus, splits.
__global__ __launch_bounds__(256) void k_prep(const float* __restrict__ zx,
                                              const float* __restrict__ conv_w,
                                              const float* __restrict__ conv_b,
                                              const float* __restrict__ dt_bias,
                                              const float* __restrict__ A_log,
                                              float* __restrict__ xs,
                                              __bf16* __restrict__ Xd,
                                              __bf16* __restrict__ Bb,
                                              __bf16* __restrict__ Cb,
                                              float* __restrict__ Ab,
                                              int dir) {
  __shared__ float sdt[NHEADS];
  int t = blockIdx.x, b = blockIdx.y, tid = threadIdx.x;
  size_t rowD = (size_t)(b * SEQL + t);

  if (tid < NHEADS) {
    int lm = memidx(t, dir);
    float raw = zx[((size_t)b * SEQL + lm) * DINPROJ + (2 * DINNER + 2 * DSTATE) + tid]
                + dt_bias[tid];
    float dt = softplusf(raw);
    sdt[tid] = dt;
    Ab[rowD * NHEADS + tid] = dt * (-expf(A_log[tid]));
  }
  __syncthreads();

  for (int ch = tid; ch < CONVDIM; ch += 256) {
    float acc = conv_b[ch];
#pragma unroll
    for (int j = 0; j < DCONV; ++j) {
      int tt = t - (DCONV - 1) + j;
      if (tt >= 0) {
        int lm = memidx(tt, dir);
        acc += conv_w[ch * DCONV + j] *
               zx[((size_t)b * SEQL + lm) * DINPROJ + DINNER + ch];
      }
    }
    float v = siluf(acc);
    if (ch < DINNER) {
      xs[rowD * DINNER + ch] = v;
      Xd[rowD * DINNER + ch] = (__bf16)(v * sdt[ch >> 6]);
    } else if (ch < DINNER + DSTATE) {
      Bb[rowD * DSTATE + (ch - DINNER)] = (__bf16)v;
    } else {
      Cb[rowD * DSTATE + (ch - DINNER - DSTATE)] = (__bf16)v;
    }
  }
}

// Per (h, c, b): intra-chunk WMMA (scores, Y_diag, chunk states).
// The 128x64 strided X tile is fetched by the Tensor Data Mover, overlapping the
// S = C x B^T WMMA phase; it is only waited on right before the Y step.
__global__ __launch_bounds__(256) void k_chunk(const __bf16* __restrict__ Bb,
                                               const __bf16* __restrict__ Cb,
                                               const __bf16* __restrict__ Xd,
                                               const float* __restrict__ Ab,
                                               float* __restrict__ Ybuf,
                                               float* __restrict__ Sbuf,
                                               float* __restrict__ ALbuf) {
  __shared__ __bf16 sS[QCHUNK * QCHUNK];     // masked decayed scores (32KB)
  __shared__ __bf16 sX[QCHUNK * HEADDIM];    // X*dt tile (16KB)
  __shared__ float  sAb[QCHUNK], sAc[QCHUNK], sW[QCHUNK];

  int h = blockIdx.x, c = blockIdx.y, b = blockIdx.z;
  int tid = threadIdx.x, lane = tid & 31, wave = tid >> 5;
  int lr = lane & 15, khalf = lane >> 4;
  int t0 = c * QCHUNK;

  const __bf16* Cg = Cb + ((size_t)b * SEQL + t0) * DSTATE;
  const __bf16* Bg = Bb + ((size_t)b * SEQL + t0) * DSTATE;
  const __bf16* Xg = Xd + ((size_t)b * SEQL + t0) * DINNER + (size_t)h * HEADDIM;

#if HAVE_TDM
  if (wave == 0) {
    unsigned long long ga = (unsigned long long)(const void*)Xg;
    unsigned lds_off =
        (unsigned)(size_t)(__attribute__((address_space(3))) __bf16*)&sX[0];
    // D# group0: count=1, lds_addr, global_addr[56:0], type=2 (image)
    u32x4 g0 = {1u, lds_off, (unsigned)ga,
                (unsigned)((ga >> 32) & 0x1FFFFFFu) | (2u << 30)};
    // D# group1: data_size=2B; tensor 64x128 tile 64x128; dim0 stride 2048 elems
    i32x8 g1 = {(int)(1u << 16),          // data_size=1 -> 2 bytes
                (int)(64u << 16),         // tensor_dim0 = 64
                (int)(128u << 16),        // tensor_dim1 = 128
                (int)(64u << 16),         // tile_dim0 = 64
                128,                      // tile_dim1 = 128, tile_dim2 = 0
                2048,                     // tensor_dim0_stride lo32
                0, 0};
    i32x4 z4 = {0, 0, 0, 0};
#if __clang_major__ >= 23
    i32x8 z8 = {0, 0, 0, 0, 0, 0, 0, 0};
    __builtin_amdgcn_tensor_load_to_lds(g0, g1, z4, z4, z8, 0);
#else
    __builtin_amdgcn_tensor_load_to_lds(g0, g1, z4, z4, 0);
#endif
  }
#else
  for (int i = tid; i < QCHUNK * HEADDIM; i += 256) {
    int l = i >> 6, p = i & 63;
    sX[i] = Xg[(size_t)l * DINNER + p];
  }
#endif

  if (tid < QCHUNK) sAb[tid] = Ab[((size_t)b * SEQL + t0 + tid) * NHEADS + h];
  __syncthreads();
  if (tid == 0) {
    float s = 0.0f;
    for (int l = 0; l < QCHUNK; ++l) { s += sAb[l]; sAc[l] = s; }
  }
  __syncthreads();
  float Alast = sAc[QCHUNK - 1];
  if (tid < QCHUNK) sW[tid] = expf(Alast - sAc[tid]);
  __syncthreads();

  // ---- S = C x B^T (128x128, K=128), decayed causal mask -> sS (bf16)
  {
    int wm = wave & 3, wn = wave >> 2;
    floatx8 acc[2][4] = {};
    for (int k0 = 0; k0 < DSTATE; k0 += 32) {
      bf16x16 af[2];
#pragma unroll
      for (int mi = 0; mi < 2; ++mi)
        af[mi] = frag_a_rowmajor(Cg, DSTATE, wm * 32 + mi * 16 + lr, khalf, k0);
#pragma unroll
      for (int ni = 0; ni < 4; ++ni) {
        bf16x16 bf = frag_b_nmajor(Bg, DSTATE, wn * 64 + ni * 16 + lr, khalf, k0);
#pragma unroll
        for (int mi = 0; mi < 2; ++mi) acc[mi][ni] = wmma_bf16(af[mi], bf, acc[mi][ni]);
      }
    }
#pragma unroll
    for (int mi = 0; mi < 2; ++mi)
#pragma unroll
      for (int ni = 0; ni < 4; ++ni)
#pragma unroll
        for (int r = 0; r < 8; ++r) {
          int l = wm * 32 + mi * 16 + r + 8 * khalf;
          int s = wn * 64 + ni * 16 + lr;
          float v = (s <= l) ? acc[mi][ni][r] * expf(sAc[l] - sAc[s]) : 0.0f;
          sS[l * QCHUNK + s] = (__bf16)v;
        }
  }
#if HAVE_TDM
  if (wave == 0) wait_tensor0();
#endif
  __syncthreads();

  // ---- Y_diag = S x X (128x64, K=128) -> global
  {
    int wm = wave & 3, wn = wave >> 2;
    floatx8 acc[2][2] = {};
    for (int k0 = 0; k0 < QCHUNK; k0 += 32) {
      bf16x16 af[2];
#pragma unroll
      for (int mi = 0; mi < 2; ++mi)
        af[mi] = frag_a_rowmajor(sS, QCHUNK, wm * 32 + mi * 16 + lr, khalf, k0);
#pragma unroll
      for (int ni = 0; ni < 2; ++ni) {
        bf16x16 bf = frag_b_kmajor(sX, HEADDIM, wn * 32 + ni * 16 + lr, khalf, k0);
#pragma unroll
        for (int mi = 0; mi < 2; ++mi) acc[mi][ni] = wmma_bf16(af[mi], bf, acc[mi][ni]);
      }
    }
#pragma unroll
    for (int mi = 0; mi < 2; ++mi)
#pragma unroll
      for (int ni = 0; ni < 2; ++ni)
#pragma unroll
        for (int r = 0; r < 8; ++r) {
          int l = wm * 32 + mi * 16 + r + 8 * khalf;
          int p = wn * 32 + ni * 16 + lr;
          Ybuf[(((size_t)b * SEQL + t0 + l) * NHEADS + h) * HEADDIM + p] = acc[mi][ni][r];
        }
  }
  __syncthreads();

  // ---- weight X by exp(Alast - Acum[l]) in place
  for (int i = tid; i < QCHUNK * HEADDIM; i += 256) {
    int l = i >> 6;
    sX[i] = (__bf16)((float)sX[i] * sW[l]);
  }
  __syncthreads();

  // ---- states = (Xw)^T x B (64x128, K=128) -> global
  {
    int wm = wave & 1, wn = wave >> 1;
    floatx8 acc[2][2] = {};
    for (int k0 = 0; k0 < QCHUNK; k0 += 32) {
      bf16x16 af[2];
#pragma unroll
      for (int mi = 0; mi < 2; ++mi)
        af[mi] = frag_a_kmajor(sX, HEADDIM, wm * 32 + mi * 16 + lr, khalf, k0);
#pragma unroll
      for (int ni = 0; ni < 2; ++ni) {
        bf16x16 bf = frag_b_kmajor(Bg, DSTATE, wn * 32 + ni * 16 + lr, khalf, k0);
#pragma unroll
        for (int mi = 0; mi < 2; ++mi) acc[mi][ni] = wmma_bf16(af[mi], bf, acc[mi][ni]);
      }
    }
    size_t base = ((((size_t)b * NCHUNK + c) * NHEADS + h)) * (HEADDIM * DSTATE);
#pragma unroll
    for (int mi = 0; mi < 2; ++mi)
#pragma unroll
      for (int ni = 0; ni < 2; ++ni)
#pragma unroll
        for (int r = 0; r < 8; ++r) {
          int p = wm * 32 + mi * 16 + r + 8 * khalf;
          int n = wn * 32 + ni * 16 + lr;
          Sbuf[base + (size_t)p * DSTATE + n] = acc[mi][ni][r];
        }
  }
  if (tid == 0) ALbuf[((size_t)b * NHEADS + h) * NCHUNK + c] = Alast;
}

// Per (b,h): sequential inter-chunk state recurrence.
__global__ __launch_bounds__(256) void k_scan(const float* __restrict__ Sbuf,
                                              const float* __restrict__ ALbuf,
                                              __bf16* __restrict__ Sin) {
  int h = blockIdx.x, b = blockIdx.y, tid = threadIdx.x;
  float carry[32];
#pragma unroll
  for (int i = 0; i < 32; ++i) carry[i] = 0.0f;
  for (int c = 0; c < NCHUNK; ++c) {
    size_t base = (((size_t)b * NCHUNK + c) * NHEADS + h) * (HEADDIM * DSTATE);
#pragma unroll
    for (int i = 0; i < 32; ++i) Sin[base + tid * 32 + i] = (__bf16)carry[i];
    float decay = expf(ALbuf[((size_t)b * NHEADS + h) * NCHUNK + c]);
#pragma unroll
    for (int i = 0; i < 32; ++i)
      carry[i] = decay * carry[i] + Sbuf[base + tid * 32 + i];
  }
}

// Per (h,c,b): Yo = C x Sin^T (128x64, K=128), fuse Y_diag + D skip + SiLU(z) gate.
__global__ __launch_bounds__(256) void k_yo(const __bf16* __restrict__ Cb,
                                            const __bf16* __restrict__ Sin,
                                            const float* __restrict__ Ab,
                                            const float* __restrict__ Ybuf,
                                            const float* __restrict__ xs,
                                            const float* __restrict__ zx,
                                            const float* __restrict__ Dskip,
                                            float* __restrict__ yg,
                                            int dir) {
  __shared__ __bf16 sSin[HEADDIM * DSTATE];    // [p][n] (16KB)
  __shared__ float  sAb[QCHUNK], sAc[QCHUNK];

  int h = blockIdx.x, c = blockIdx.y, b = blockIdx.z;
  int tid = threadIdx.x, lane = tid & 31, wave = tid >> 5;
  int lr = lane & 15, khalf = lane >> 4;
  int t0 = c * QCHUNK;

  const __bf16* Cg = Cb + ((size_t)b * SEQL + t0) * DSTATE;
  const __bf16* Sg = Sin + (((size_t)b * NCHUNK + c) * NHEADS + h) * (HEADDIM * DSTATE);
  for (int i = tid; i < HEADDIM * DSTATE; i += 256) sSin[i] = Sg[i];
  if (tid < QCHUNK) sAb[tid] = Ab[((size_t)b * SEQL + t0 + tid) * NHEADS + h];
  __syncthreads();
  if (tid == 0) {
    float s = 0.0f;
    for (int l = 0; l < QCHUNK; ++l) { s += sAb[l]; sAc[l] = s; }
  }
  __syncthreads();

  float Dk = Dskip[h];
  int wm = wave & 3, wn = wave >> 2;
  floatx8 acc[2][2] = {};
  for (int k0 = 0; k0 < DSTATE; k0 += 32) {
    bf16x16 af[2];
#pragma unroll
    for (int mi = 0; mi < 2; ++mi)
      af[mi] = frag_a_rowmajor(Cg, DSTATE, wm * 32 + mi * 16 + lr, khalf, k0);
#pragma unroll
    for (int ni = 0; ni < 2; ++ni) {
      bf16x16 bf = frag_b_nmajor(sSin, DSTATE, wn * 32 + ni * 16 + lr, khalf, k0);
#pragma unroll
      for (int mi = 0; mi < 2; ++mi) acc[mi][ni] = wmma_bf16(af[mi], bf, acc[mi][ni]);
    }
  }
#pragma unroll
  for (int mi = 0; mi < 2; ++mi)
#pragma unroll
    for (int ni = 0; ni < 2; ++ni)
#pragma unroll
      for (int r = 0; r < 8; ++r) {
        int l = wm * 32 + mi * 16 + r + 8 * khalf;
        int p = wn * 32 + ni * 16 + lr;
        int t = t0 + l;
        int ch = h * HEADDIM + p;
        size_t rowD = (size_t)b * SEQL + t;
        float yd = Ybuf[rowD * DINNER + ch];
        float yo = expf(sAc[l]) * acc[mi][ni][r];
        float xv = xs[rowD * DINNER + ch];
        float zraw = zx[((size_t)b * SEQL + memidx(t, dir)) * DINPROJ + ch];
        yg[rowD * DINNER + ch] = (yd + yo + Dk * xv) * siluf(zraw);
      }
}

// Per (b,t): gated RMSNorm and accumulate time-aligned into combined buffer.
__global__ __launch_bounds__(256) void k_rms(const float* __restrict__ yg,
                                             const float* __restrict__ norm_w,
                                             float* __restrict__ yc, int dir) {
  __shared__ float red[256];
  int t = blockIdx.x, b = blockIdx.y, tid = threadIdx.x;
  size_t rowD = (size_t)b * SEQL + t;
  float part = 0.0f;
  float v[8];
#pragma unroll
  for (int i = 0; i < 8; ++i) {
    v[i] = yg[rowD * DINNER + tid + i * 256];
    part += v[i] * v[i];
  }
  red[tid] = part;
  __syncthreads();
  for (int s = 128; s > 0; s >>= 1) {
    if (tid < s) red[tid] += red[tid + s];
    __syncthreads();
  }
  float rs = rsqrtf(red[0] / (float)DINNER + 1e-5f);
  size_t rowM = (size_t)b * SEQL + memidx(t, dir);
#pragma unroll
  for (int i = 0; i < 8; ++i) {
    int ch = tid + i * 256;
    float o = v[i] * rs * norm_w[ch];
    if (dir == 0) yc[rowM * DINNER + ch] = o;
    else          yc[rowM * DINNER + ch] += o;
  }
}

// ---------------- host orchestration ----------------
extern "C" void kernel_launch(void* const* d_in, const int* in_sizes, int n_in,
                              void* d_out, int out_size, void* d_ws, size_t ws_size,
                              hipStream_t stream) {
  (void)in_sizes; (void)n_in; (void)out_size; (void)ws_size;
  const float* x      = (const float*)d_in[0];
  const float* W_in   = (const float*)d_in[1];
  const float* W_out  = (const float*)d_in[2];
  const float* conv_w[2]  = {(const float*)d_in[3], (const float*)d_in[9]};
  const float* conv_b[2]  = {(const float*)d_in[4], (const float*)d_in[10]};
  const float* dt_bias[2] = {(const float*)d_in[5], (const float*)d_in[11]};
  const float* A_log[2]   = {(const float*)d_in[6], (const float*)d_in[12]};
  const float* Dskip[2]   = {(const float*)d_in[7], (const float*)d_in[13]};
  const float* norm_w[2]  = {(const float*)d_in[8], (const float*)d_in[14]};
  float* out = (float*)d_out;

  char* ws = (char*)d_ws;
  __bf16* win_bf  = (__bf16*)(ws + OFF_WIN_BF);
  __bf16* wout_bf = (__bf16*)(ws + OFF_WOUT_BF);
  float*  zx      = (float*) (ws + OFF_ZX);
  float*  xs      = (float*) (ws + OFF_XS);
  __bf16* Xd      = (__bf16*)(ws + OFF_XD);
  __bf16* Bb      = (__bf16*)(ws + OFF_BB);
  __bf16* Cbuf    = (__bf16*)(ws + OFF_CB);
  float*  Ab      = (float*) (ws + OFF_AB);
  float*  Ybuf    = (float*) (ws + OFF_Y);
  float*  Sbuf    = (float*) (ws + OFF_ST);
  __bf16* Sin     = (__bf16*)(ws + OFF_SIN);
  float*  ALbuf   = (float*) (ws + OFF_AL);
  float*  yg      = (float*) (ws + OFF_YG);
  float*  yc      = (float*) (ws + OFF_YC);
  __bf16* xbf     = (__bf16*)(ws + OFF_XBF);
  __bf16* ycbf    = (__bf16*)(ws + OFF_YCBF);

  // 1) weights + activations -> bf16
  {
    int n1 = DINPROJ * DMODEL;
    k_cvt_bf16<<<(n1 + 255) / 256, 256, 0, stream>>>(W_in, win_bf, n1);
    int n2 = DMODEL * DINNER;
    k_cvt_bf16<<<(n2 + 255) / 256, 256, 0, stream>>>(W_out, wout_bf, n2);
    int n3 = BL * DMODEL;
    k_cvt_bf16<<<(n3 + 255) / 256, 256, 0, stream>>>(x, xbf, n3);
  }
  // 2) zxbcdt = x @ W_in^T  (4096 x 4384, K=1024)
  {
    dim3 grid((DINPROJ + 127) / 128, BL / 128);
    k_gemm_bf16<<<grid, 256, 0, stream>>>(xbf, win_bf, zx, BL, DINPROJ, DMODEL);
  }
  // 3) both directions
  for (int dir = 0; dir < 2; ++dir) {
    k_prep<<<dim3(SEQL, BATCH), 256, 0, stream>>>(zx, conv_w[dir], conv_b[dir],
                                                  dt_bias[dir], A_log[dir],
                                                  xs, Xd, Bb, Cbuf, Ab, dir);
    k_chunk<<<dim3(NHEADS, NCHUNK, BATCH), 256, 0, stream>>>(Bb, Cbuf, Xd, Ab,
                                                             Ybuf, Sbuf, ALbuf);
    k_scan<<<dim3(NHEADS, BATCH), 256, 0, stream>>>(Sbuf, ALbuf, Sin);
    k_yo<<<dim3(NHEADS, NCHUNK, BATCH), 256, 0, stream>>>(Cbuf, Sin, Ab, Ybuf, xs, zx,
                                                          Dskip[dir], yg, dir);
    k_rms<<<dim3(SEQL, BATCH), 256, 0, stream>>>(yg, norm_w[dir], yc, dir);
  }
  // 4) out = (y_f + y_r) @ W_out^T  (4096 x 1024, K=2048)
  {
    int n4 = BL * DINNER;
    k_cvt_bf16<<<(n4 + 255) / 256, 256, 0, stream>>>(yc, ycbf, n4);
    dim3 grid(DMODEL / 128, BL / 128);
    k_gemm_bf16<<<grid, 256, 0, stream>>>(ycbf, wout_bf, out, BL, DMODEL, DINNER);
  }
}
